// DSTP_rnn_20675972563501
// MI455X (gfx1250) — matrix operand order repaired and checked by reading, not compiled, not run on hardware
//
#include <hip/hip_runtime.h>

// ---------------------------------------------------------------------------
// DSTP-RNN for MI455X (gfx1250): per-workgroup 16-row batch tile, bf16 WMMA
// (v_wmma_f32_16x16x32_bf16) for all gate / attention GEMMs, fp32 accumulate,
// LSTM cell state in registers, hidden states mirrored to LDS as bf16.
// Weights repacked once into WMMA B-fragment layout (bf16) in d_ws.
// K-chunk loops are kept rolled (#pragma unroll 1) to bound register pressure
// (fully unrolling them caused B-fragment spills to scratch).
// ---------------------------------------------------------------------------

typedef __attribute__((ext_vector_type(8)))  float  v8f;
typedef __attribute__((ext_vector_type(16))) __bf16 v16bf;

#define T15 15
#define DD  81
#define HE  256

// ---- ws fragment offsets (ushort elements). Each fragment = 32x16 bf16 = 512
static const size_t OFF_L1WIH = 0;         // KC=3,  NT=64
static const size_t OFF_L1WHH = 98304;     // KC=8,  NT=64
static const size_t OFF_L2WIH = 360448;    // KC=3,  NT=64
static const size_t OFF_L2WHH = 458752;    // KC=8,  NT=64
static const size_t OFF_DLWHH = 720896;    // KC=8,  NT=64
static const size_t OFF_W1X   = 983040;    // KC=8,  NT=16
static const size_t OFF_W1D   = 1048576;   // KC=8,  NT=16
static const size_t OFF_W1C   = 1114112;   // KC=8,  NT=16
// total = 1179648 ushorts = 2359296 bytes of d_ws

// ---- LDS layout (bytes)
static const unsigned LDS_XENC  = 0;        // bf16 [15][16][256] = 122880
static const unsigned LDS_XPROJ = 122880;   // bf16 [15][16][256] = 122880
static const unsigned LDS_L1B   = 245760;   // f32[1024]
static const unsigned LDS_L2B   = 249856;
static const unsigned LDS_DLB   = 253952;
static const unsigned LDS_DLW   = 258048;   // dl_wih vector f32[1024]
static const unsigned LDS_DA1B  = 262144;   // f32[256]
static const unsigned LDS_DA2W  = 263168;   // f32[256]
static const unsigned LDS_FCW   = 264192;   // f32[257] (pad 1088B)
static const unsigned LDS_FFW   = 265280;   // f32[512]
static const unsigned LDS_U     = 267328;   // union scratch region
// encoder view of U:
//   a1 f32[16*96]@+0, a2@+6144, xt1 u16[16*96]@+12288, xt2@+15360,
//   h1 u16[16*256]@+18432, hs@+26624                          (34816 B)
// decoder view of U:
//   dd f32[16*256]@+0, ctx f32@+16384, dB u16@+32768, cB u16@+40960,
//   score f32[16*16]@+49152, beta@+50176, yt f32[16]@+51200   (51264 B)
static const unsigned SHMEM_BYTES = LDS_U + 51264;  // 318592 <= 320KB

// ---------------------------------------------------------------------------
__device__ __forceinline__ unsigned short f2bf(float f) {
  union { float f; unsigned int u; } x; x.f = f;
  unsigned int u = x.u;
  return (unsigned short)((u + 0x7fffu + ((u >> 16) & 1u)) >> 16);
}
__device__ __forceinline__ float bf2f(unsigned short h) {
  union { unsigned int u; float f; } x; x.u = ((unsigned int)h) << 16;
  return x.f;
}
__device__ __forceinline__ float sigm(float x) { return 1.0f / (1.0f + expf(-x)); }
__device__ __forceinline__ v8f splat8(float x) {
  v8f v;
#pragma unroll
  for (int i = 0; i < 8; ++i) v[i] = x;
  return v;
}

// A-fragment (16x32 bf16, M x K) from LDS row-major [16][strideHalves]:
// lanes 0-15: halves 0..7 -> K0..K0+7 ; halves 8..15 -> K0+16..K0+23 (K0=kc*32)
// lanes 16-31: +8 on base K.
__device__ __forceinline__ v16bf loadA_lds(const unsigned short* base,
                                           int strideHalves, int kc, int lane) {
  int M  = lane & 15;
  int k0 = kc * 32 + ((lane >> 4) << 3);
  const unsigned short* p = base + M * strideHalves + k0;
  union { uint4 q[2]; v16bf v; } u;
  u.q[0] = *(const uint4*)(p);
  u.q[1] = *(const uint4*)(p + 16);
  return u.v;
}

// B-fragment (32x16 bf16, K x N), pre-packed lane-major in d_ws: lane stores
// its 16 halves contiguously (lanes 0-15: K=0..7,16..23,N=lane; 16-31: K=8..15,24..31).
__device__ __forceinline__ v16bf loadB_ws(const unsigned short* frag, int lane) {
  const unsigned short* p = frag + lane * 16;
  union { uint4 q[2]; v16bf v; } u;
  u.q[0] = *(const uint4*)(p);
  u.q[1] = *(const uint4*)(p + 8);
  return u.v;
}

__device__ __forceinline__ v8f wmma_bf16(v16bf a, v16bf b, v8f c) {
  return __builtin_amdgcn_wmma_f32_16x16x32_bf16(false, a, false, b,
                                                 (short)0, c, false, false);
}

// full LSTM gate GEMM: acc[g][s] (g=i,f,g,o ; s=two 16-wide N subtiles)
// K-chunk loops rolled to bound live B-fragments (8 in flight per iteration).
__device__ __forceinline__ void lstm_gemm(v8f acc[4][2],
    const unsigned short* xt96, size_t offWih,
    const unsigned short* h256, size_t offWhh,
    const unsigned short* wsB, int wv, int lane) {
#pragma unroll 1
  for (int kc = 0; kc < 3; ++kc) {
    v16bf a = loadA_lds(xt96, 96, kc, lane);
#pragma unroll
    for (int g = 0; g < 4; ++g)
#pragma unroll
      for (int sj = 0; sj < 2; ++sj)
        acc[g][sj] = wmma_bf16(a,
            loadB_ws(wsB + offWih + (size_t)(kc * 64 + g * 16 + wv * 2 + sj) * 512, lane),
            acc[g][sj]);
  }
#pragma unroll 1
  for (int kc = 0; kc < 8; ++kc) {
    v16bf a = loadA_lds(h256, 256, kc, lane);
#pragma unroll
    for (int g = 0; g < 4; ++g)
#pragma unroll
      for (int sj = 0; sj < 2; ++sj)
        acc[g][sj] = wmma_bf16(a,
            loadB_ws(wsB + offWhh + (size_t)(kc * 64 + g * 16 + wv * 2 + sj) * 512, lane),
            acc[g][sj]);
  }
}

// ---------------------------------------------------------------------------
// Weight repack: B[k][n] = W[n][col_off + k] -> bf16 WMMA B-fragments.
__global__ void __launch_bounds__(256) dstp_repack(
    const float* __restrict__ W, unsigned short* __restrict__ dst,
    int NT, int KC, int K_in, int ldw, int col_off) {
  int f    = blockIdx.x * 8 + (threadIdx.x >> 5);
  int lane = threadIdx.x & 31;
  if (f >= NT * KC) return;
  int kc = f / NT, nt = f % NT;
  int N  = nt * 16 + (lane & 15);
  int Kb = kc * 32 + ((lane >> 4) << 3);
  unsigned short* o = dst + (size_t)f * 512 + lane * 16;
#pragma unroll
  for (int e = 0; e < 16; ++e) {
    int K = Kb + e + ((e >> 3) << 3);
    float v = (K < K_in) ? W[(size_t)N * ldw + col_off + K] : 0.0f;
    o[e] = f2bf(v);
  }
}

// ---------------------------------------------------------------------------
__global__ void __launch_bounds__(256, 1) dstp_main(
    const float* __restrict__ X,      const float* __restrict__ y_prev,
    const float* __restrict__ e1w,    const float* __restrict__ e2w,
    const float* __restrict__ l1_b,   const float* __restrict__ l2_b,
    const float* __restrict__ da1b,   const float* __restrict__ da2w,
    const float* __restrict__ dlwih,  const float* __restrict__ dl_b,
    const float* __restrict__ fcw,    const float* __restrict__ fcb,
    const float* __restrict__ ffw,    const float* __restrict__ ffb,
    const unsigned short* __restrict__ wsB, float* __restrict__ out) {
  extern __shared__ char smem[];
  const int tid  = threadIdx.x;
  const int lane = tid & 31;
  const int wv   = tid >> 5;
  const int nl   = lane & 15;
  const int Ml   = (lane < 16) ? 0 : 8;      // C/D layout: M = Ml + vgpr_r
  const int n0   = wv * 32;                  // this wave's hidden slice
  const int row0 = blockIdx.x * 16;

  unsigned short* Xenc  = (unsigned short*)(smem + LDS_XENC);
  unsigned short* Xproj = (unsigned short*)(smem + LDS_XPROJ);
  float* s_l1b  = (float*)(smem + LDS_L1B);
  float* s_l2b  = (float*)(smem + LDS_L2B);
  float* s_dlb  = (float*)(smem + LDS_DLB);
  float* s_dlw  = (float*)(smem + LDS_DLW);
  float* s_da1b = (float*)(smem + LDS_DA1B);
  float* s_da2w = (float*)(smem + LDS_DA2W);
  float* s_fcw  = (float*)(smem + LDS_FCW);
  float* s_ffw  = (float*)(smem + LDS_FFW);
  // encoder view
  float*          s_a1  = (float*)(smem + LDS_U);
  float*          s_a2  = (float*)(smem + LDS_U + 6144);
  unsigned short* s_xt1 = (unsigned short*)(smem + LDS_U + 12288);
  unsigned short* s_xt2 = (unsigned short*)(smem + LDS_U + 15360);
  unsigned short* s_h1  = (unsigned short*)(smem + LDS_U + 18432);
  unsigned short* s_hs  = (unsigned short*)(smem + LDS_U + 26624);
  // decoder view (aliases encoder view; only valid after encoder completes)
  float*          s_dd  = (float*)(smem + LDS_U);
  float*          s_ctx = (float*)(smem + LDS_U + 16384);
  unsigned short* s_dB  = (unsigned short*)(smem + LDS_U + 32768);
  unsigned short* s_cB  = (unsigned short*)(smem + LDS_U + 40960);
  float*          s_sc  = (float*)(smem + LDS_U + 49152);
  float*          s_be  = (float*)(smem + LDS_U + 50176);
  float*          s_yt  = (float*)(smem + LDS_U + 51200);

  // ---- preload small weights, zero state buffers -------------------------
  for (int i = tid; i < 1024; i += 256) {
    s_l1b[i] = l1_b[i]; s_l2b[i] = l2_b[i]; s_dlb[i] = dl_b[i]; s_dlw[i] = dlwih[i];
  }
  if (tid < 256) { s_da1b[tid] = da1b[tid]; s_da2w[tid] = da2w[tid]; }
  for (int i = tid; i < 257; i += 256) s_fcw[i] = fcw[i];
  for (int i = tid; i < 512; i += 256) s_ffw[i] = ffw[i];
  for (int i = tid; i < 16 * 96; i += 256) { s_xt1[i] = 0; s_xt2[i] = 0; }
  for (int i = tid; i < 16 * 256; i += 256) { s_h1[i] = 0; s_hs[i] = 0; }

  // ---- static attention weights: a1 = softmax_d(X·w1x), a2 = softmax_d(X·w2x)
  // (the recurrent terms and scalar biases are constant along the softmax axis)
  for (int idx = tid; idx < 16 * DD; idx += 256) {
    int r = idx / DD, d = idx % DD;
    const float* xr = X + (size_t)(row0 + r) * (T15 * DD) + d;
    float p1 = 0.f, p2 = 0.f;
#pragma unroll
    for (int t = 0; t < T15; ++t) {
      float xv = xr[t * DD];
      p1 += xv * e1w[512 + t];
      p2 += xv * e2w[512 + t];
    }
    s_a1[r * 96 + d] = p1;
    s_a2[r * 96 + d] = p2;
  }
  __syncthreads();
  if (tid < 32) {                       // 16 rows x 2 arrays
    float* a = (tid < 16) ? s_a1 : s_a2;
    int r = tid & 15;
    float m = -1e30f;
    for (int d = 0; d < DD; ++d) m = fmaxf(m, a[r * 96 + d]);
    float se = 0.f;
    for (int d = 0; d < DD; ++d) { float e = expf(a[r * 96 + d] - m); a[r * 96 + d] = e; se += e; }
    float inv = 1.0f / se;
    for (int d = 0; d < DD; ++d) a[r * 96 + d] *= inv;
  }

  // ---- encoder -----------------------------------------------------------
  float c1c[2][8] = {};   // LSTM1 cell, in registers
  float c2c[2][8] = {};   // LSTM2 cell
#pragma unroll 1
  for (int t = 0; t < T15; ++t) {
    __syncthreads();
    // x_tilde = a1*xt ; x_tilde2 = a2*x_tilde  (bf16 A-buffers, zero-padded to 96)
    for (int idx = tid; idx < 16 * DD; idx += 256) {
      int r = idx / DD, d = idx % DD;
      float xv = X[(size_t)(row0 + r) * (T15 * DD) + t * DD + d];
      float x1 = s_a1[r * 96 + d] * xv;
      s_xt1[r * 96 + d] = f2bf(x1);
      s_xt2[r * 96 + d] = f2bf(s_a2[r * 96 + d] * x1);
    }
    __syncthreads();

    v8f acc[4][2];
#pragma unroll
    for (int g = 0; g < 4; ++g)
#pragma unroll
      for (int sj = 0; sj < 2; ++sj)
        acc[g][sj] = splat8(s_l1b[g * 256 + n0 + sj * 16 + nl]);
    lstm_gemm(acc, s_xt1, OFF_L1WIH, s_h1, OFF_L1WHH, wsB, wv, lane);
    __syncthreads();                         // all waves done reading h1_old
#pragma unroll
    for (int sj = 0; sj < 2; ++sj)
#pragma unroll
      for (int r = 0; r < 8; ++r) {
        float cn = sigm(acc[1][sj][r]) * c1c[sj][r] +
                   sigm(acc[0][sj][r]) * tanhf(acc[2][sj][r]);
        c1c[sj][r] = cn;
        s_h1[(Ml + r) * 256 + n0 + sj * 16 + nl] = f2bf(sigm(acc[3][sj][r]) * tanhf(cn));
      }

#pragma unroll
    for (int g = 0; g < 4; ++g)
#pragma unroll
      for (int sj = 0; sj < 2; ++sj)
        acc[g][sj] = splat8(s_l2b[g * 256 + n0 + sj * 16 + nl]);
    lstm_gemm(acc, s_xt2, OFF_L2WIH, s_hs, OFF_L2WHH, wsB, wv, lane);
    __syncthreads();                         // all waves done reading hs_old
#pragma unroll
    for (int sj = 0; sj < 2; ++sj)
#pragma unroll
      for (int r = 0; r < 8; ++r) {
        float cn = sigm(acc[1][sj][r]) * c2c[sj][r] +
                   sigm(acc[0][sj][r]) * tanhf(acc[2][sj][r]);
        c2c[sj][r] = cn;
        unsigned short hb = f2bf(sigm(acc[3][sj][r]) * tanhf(cn));
        int M = Ml + r, N = n0 + sj * 16 + nl;
        s_hs[M * 256 + N] = hb;
        Xenc[(t * 16 + M) * 256 + N] = hb;   // X_encoded
      }
  }
  __syncthreads();                           // encoder scratch now dead

  // ---- decoder setup -----------------------------------------------------
  for (int i = tid; i < 16 * 256; i += 256) { s_dB[i] = 0; s_cB[i] = 0; }
  // xenc_proj = X_encoded @ W1x^T + d_a1_b  (240 M-tiles*N-tiles over 8 waves)
#pragma unroll 1
  for (int tt = wv; tt < 240; tt += 8) {
    int mt = tt >> 4, nt = tt & 15;
    v8f a1c = splat8(s_da1b[nt * 16 + nl]);
#pragma unroll 1
    for (int kc = 0; kc < 8; ++kc)
      a1c = wmma_bf16(loadA_lds(Xenc + mt * 16 * 256, 256, kc, lane),
                      loadB_ws(wsB + OFF_W1X + (size_t)(kc * 16 + nt) * 512, lane), a1c);
#pragma unroll
    for (int r = 0; r < 8; ++r)
      Xproj[(mt * 16 + Ml + r) * 256 + nt * 16 + nl] = f2bf(a1c[r]);
  }
  __syncthreads();

  float cdc[2][8] = {};                      // decoder LSTM cell
#pragma unroll 1
  for (int t = 0; t < T15; ++t) {
    __syncthreads();
    // dd = d @ W1d^T + c @ W1c^T
    {
      v8f ac[2] = { splat8(0.f), splat8(0.f) };
#pragma unroll 1
      for (int kc = 0; kc < 8; ++kc) {
        v16bf a = loadA_lds(s_dB, 256, kc, lane);
#pragma unroll
        for (int sj = 0; sj < 2; ++sj)
          ac[sj] = wmma_bf16(a, loadB_ws(wsB + OFF_W1D + (size_t)(kc * 16 + wv * 2 + sj) * 512, lane), ac[sj]);
      }
#pragma unroll 1
      for (int kc = 0; kc < 8; ++kc) {
        v16bf a = loadA_lds(s_cB, 256, kc, lane);
#pragma unroll
        for (int sj = 0; sj < 2; ++sj)
          ac[sj] = wmma_bf16(a, loadB_ws(wsB + OFF_W1C + (size_t)(kc * 16 + wv * 2 + sj) * 512, lane), ac[sj]);
      }
#pragma unroll
      for (int sj = 0; sj < 2; ++sj)
#pragma unroll
        for (int r = 0; r < 8; ++r)
          s_dd[(Ml + r) * 256 + (wv * 2 + sj) * 16 + nl] = ac[sj][r];
    }
    __syncthreads();
    // score[r][t'] = sum_n tanh(xproj + dd) * d_a2_w[n]
    if (tid < 240) {
      int r = tid / 15, tp = tid % 15;
      const unsigned short* xp = Xproj + (tp * 16 + r) * 256;
      const float* ddp = s_dd + r * 256;
      float s = 0.f;
      for (int n = 0; n < 256; ++n)
        s += tanhf(bf2f(xp[n]) + ddp[n]) * s_da2w[n];
      s_sc[r * 16 + tp] = s;
    }
    __syncthreads();
    if (tid < 16) {                           // beta = softmax_t(score)
      float m = -1e30f;
      for (int k = 0; k < T15; ++k) m = fmaxf(m, s_sc[tid * 16 + k]);
      float se = 0.f;
      for (int k = 0; k < T15; ++k) { float e = expf(s_sc[tid * 16 + k] - m); s_be[tid * 16 + k] = e; se += e; }
      float inv = 1.0f / se;
      for (int k = 0; k < T15; ++k) s_be[tid * 16 + k] *= inv;
    }
    __syncthreads();
    {                                         // ctx = beta @ X_encoded
      int r = tid >> 4, nb = (tid & 15) * 16;
#pragma unroll
      for (int k = 0; k < 16; ++k) {
        int n = nb + k;
        float a = 0.f;
#pragma unroll
        for (int tp = 0; tp < T15; ++tp)
          a += s_be[r * 16 + tp] * bf2f(Xenc[(tp * 16 + r) * 256 + n]);
        s_ctx[r * 256 + n] = a;
      }
    }
    __syncthreads();
    {                                         // y_tilde = [ctx, y]·fc_w + fc_b
      int r = tid >> 4, j = tid & 15;
      float p = 0.f;
#pragma unroll
      for (int k = 0; k < 16; ++k) p += s_ctx[r * 256 + j * 16 + k] * s_fcw[j * 16 + k];
#pragma unroll
      for (int m = 8; m; m >>= 1) p += __shfl_xor(p, m, 32);
      if (j == 0)
        s_yt[r] = p + y_prev[(size_t)(row0 + r) * T15 + t] * s_fcw[256] + fcb[0];
    }
    __syncthreads();
    // decoder LSTM: g = y_tilde*dl_wih + d@dl_whh^T + dl_b
    {
      float ytv[8];
#pragma unroll
      for (int r = 0; r < 8; ++r) ytv[r] = s_yt[Ml + r];
      v8f acc[4][2];
#pragma unroll
      for (int g = 0; g < 4; ++g)
#pragma unroll
        for (int sj = 0; sj < 2; ++sj) {
          int N = g * 256 + n0 + sj * 16 + nl;
          float bb = s_dlb[N], ww = s_dlw[N];
#pragma unroll
          for (int r = 0; r < 8; ++r) acc[g][sj][r] = bb + ytv[r] * ww;
        }
#pragma unroll 1
      for (int kc = 0; kc < 8; ++kc) {
        v16bf a = loadA_lds(s_dB, 256, kc, lane);
#pragma unroll
        for (int g = 0; g < 4; ++g)
#pragma unroll
          for (int sj = 0; sj < 2; ++sj)
            acc[g][sj] = wmma_bf16(a,
                loadB_ws(wsB + OFF_DLWHH + (size_t)(kc * 64 + g * 16 + wv * 2 + sj) * 512, lane),
                acc[g][sj]);
      }
      __syncthreads();                        // all waves done reading d_old
#pragma unroll
      for (int sj = 0; sj < 2; ++sj)
#pragma unroll
        for (int r = 0; r < 8; ++r) {
          float cn = sigm(acc[1][sj][r]) * cdc[sj][r] +
                     sigm(acc[0][sj][r]) * tanhf(acc[2][sj][r]);
          cdc[sj][r] = cn;
          int idx = (Ml + r) * 256 + n0 + sj * 16 + nl;
          s_dB[idx] = f2bf(sigm(acc[3][sj][r]) * tanhf(cn));
          s_cB[idx] = f2bf(cn);
        }
    }
  }
  __syncthreads();
  // y_pred = [d, ctx] · ff_w + ff_b
  {
    int r = tid >> 4, j = tid & 15;
    float p = 0.f;
#pragma unroll
    for (int k = 0; k < 16; ++k) {
      int n = j * 16 + k;
      p += bf2f(s_dB[r * 256 + n]) * s_ffw[n] + s_ctx[r * 256 + n] * s_ffw[256 + n];
    }
#pragma unroll
    for (int m = 8; m; m >>= 1) p += __shfl_xor(p, m, 32);
    if (j == 0) out[row0 + r] = p + ffb[0];
  }
}

// ---------------------------------------------------------------------------
extern "C" void kernel_launch(void* const* d_in, const int* in_sizes, int n_in,
                              void* d_out, int out_size, void* d_ws, size_t ws_size,
                              hipStream_t stream) {
  (void)in_sizes; (void)n_in; (void)out_size; (void)ws_size;
  const float* X      = (const float*)d_in[0];
  const float* y_prev = (const float*)d_in[1];
  const float* e1w    = (const float*)d_in[2];
  const float* e2w    = (const float*)d_in[4];
  const float* l1_wih = (const float*)d_in[6];
  const float* l1_whh = (const float*)d_in[7];
  const float* l1_b   = (const float*)d_in[8];
  const float* l2_wih = (const float*)d_in[9];
  const float* l2_whh = (const float*)d_in[10];
  const float* l2_b   = (const float*)d_in[11];
  const float* d_a1_w = (const float*)d_in[12];
  const float* d_a1_b = (const float*)d_in[13];
  const float* d_a2_w = (const float*)d_in[14];
  const float* dl_wih = (const float*)d_in[16];
  const float* dl_whh = (const float*)d_in[17];
  const float* dl_b   = (const float*)d_in[18];
  const float* fc_w   = (const float*)d_in[19];
  const float* fc_b   = (const float*)d_in[20];
  const float* ff_w   = (const float*)d_in[21];
  const float* ff_b   = (const float*)d_in[22];
  unsigned short* ws  = (unsigned short*)d_ws;
  float* out          = (float*)d_out;

  // repack all weight matrices into bf16 WMMA B-fragment layout (L2-resident)
  dstp_repack<<<24, 256, 0, stream>>>(l1_wih, ws + OFF_L1WIH, 64, 3,  81,  81, 0);
  dstp_repack<<<64, 256, 0, stream>>>(l1_whh, ws + OFF_L1WHH, 64, 8, 256, 256, 0);
  dstp_repack<<<24, 256, 0, stream>>>(l2_wih, ws + OFF_L2WIH, 64, 3,  81,  81, 0);
  dstp_repack<<<64, 256, 0, stream>>>(l2_whh, ws + OFF_L2WHH, 64, 8, 256, 256, 0);
  dstp_repack<<<64, 256, 0, stream>>>(dl_whh, ws + OFF_DLWHH, 64, 8, 256, 256, 0);
  dstp_repack<<<16, 256, 0, stream>>>(d_a1_w, ws + OFF_W1X,   16, 8, 256, 768, 512);
  dstp_repack<<<16, 256, 0, stream>>>(d_a1_w, ws + OFF_W1D,   16, 8, 256, 768, 0);
  dstp_repack<<<16, 256, 0, stream>>>(d_a1_w, ws + OFF_W1C,   16, 8, 256, 768, 256);

  // 8192 rows / 16 rows-per-WG = 512 workgroups, 8 waves each
  dstp_main<<<512, 256, SHMEM_BYTES, stream>>>(
      X, y_prev, e1w, e2w, l1_b, l2_b, d_a1_b, d_a2_w,
      dl_wih, dl_b, fc_w, fc_b, ff_w, ff_b, ws, out);
}
